// GCPN_CReM_65000035058183
// MI455X (gfx1250) — matrix operand order repaired and checked by reading, not compiled, not run on hardware
//
#include <hip/hip_runtime.h>
#include <hip/hip_bf16.h>

typedef __attribute__((ext_vector_type(16))) __bf16 v16bf;
typedef __attribute__((ext_vector_type(8)))  float  v8f;

#define WAVES 8
#define EPS_F 1e-4f

// ---------------------------------------------------------------------------
// Fused gather + concat + 3-layer MLP (bf16 WMMA, f32 accumulate) + X_states.
// One wave32 owns a 16-row tile. Weights pre-swizzled to B-fragment layout in
// LDS; A fragments built straight from global (layer0) / per-wave LDS h-buf.
// ---------------------------------------------------------------------------
__global__ __launch_bounds__(256) void mlp_wmma_kernel(
    const float* __restrict__ gemb, const float* __restrict__ cand,
    const int* __restrict__ bidx,
    const float* __restrict__ W0, const float* __restrict__ b0,
    const float* __restrict__ W1, const float* __restrict__ b1,
    const float* __restrict__ W2, const float* __restrict__ b2,
    const float* __restrict__ Wf, const float* __restrict__ bfp,
    float* __restrict__ outX, float* __restrict__ logits, int N)
{
  // 32 B-fragments (W0:16, W1:8, W2:8), each 32 lanes x 16 bf16 = 1KB
  __shared__ __align__(32) __bf16 wfrag[32 * 32 * 16];
  __shared__ __align__(32) __bf16 hbuf[WAVES][16 * 64];
  __shared__ float biasLds[3][64];
  __shared__ float wfLds[64];
  __shared__ float bfLds;

  const int tid = threadIdx.x;

  // ---- One-time per block: swizzle fp32 weights -> bf16 B-fragments in LDS.
  // B-operand layout (K x 16): lane L holds column n = nt*16 + (L&15),
  // K runs [c0,c0+8) and [c0+16,c0+24) with c0 = kt*32 + (L>=16 ? 8 : 0).
  for (int item = tid; item < 32 * 32; item += 256) {
    const int f = item >> 5, l = item & 31;
    const float* src; int kt, nt;
    if (f < 16)      { src = W0; kt = f >> 2;      nt = f & 3; }
    else if (f < 24) { src = W1; kt = (f-16) >> 2; nt = (f-16) & 3; }
    else             { src = W2; kt = (f-24) >> 2; nt = (f-24) & 3; }
    const int Kbase = kt * 32 + ((l >= 16) ? 8 : 0);
    const int n = nt * 16 + (l & 15);
#pragma unroll
    for (int e = 0; e < 16; ++e)
      wfrag[(f * 32 + l) * 16 + e] = (__bf16)src[(Kbase + (e & 7) + ((e >> 3) << 4)) * 64 + n];
  }
  if (tid < 64) {
    biasLds[0][tid] = b0[tid];
    biasLds[1][tid] = b1[tid];
    biasLds[2][tid] = b2[tid];
    wfLds[tid] = Wf[tid];
  }
  if (tid == 0) bfLds = bfp[0];
  __syncthreads();

  const int lane    = tid & 31;
  const int wave    = tid >> 5;
  const int m       = lane & 15;   // row within tile (A) / col within N-tile (B,C)
  const int halfw   = lane >> 4;
  const int laneoff = halfw * 8;   // K sub-offset for 16-bit A layout
  __bf16* h = &hbuf[wave][0];

  const long nTiles = (long)N >> 4;
  for (long t = (long)blockIdx.x * WAVES + wave; t < nTiles;
       t += (long)gridDim.x * WAVES) {
    const long r0 = t << 4;
    const long r  = r0 + m;
    const int  bi = bidx[r];
    const float* grow = gemb + (long)bi * 64;
    const float* crow = cand + r * 64;

    // ---- X_states[r] = concat(g_emb[bi], cand[r]); lane covers 64-col half.
    {
      const float4* s4 = (const float4*)(halfw ? crow : grow);
      float4* d4 = (float4*)(outX + r * 128 + halfw * 64);
#pragma unroll
      for (int q = 0; q < 16; ++q) d4[q] = s4[q];
    }

    // ---- Layer 0: [16x128] @ [128x64].  A layout per 16-bit 16x32 spec:
    // lane holds K runs [c0,c0+8) and [c0+16,c0+24), c0 = kt*32 + laneoff.
    v16bf a0[4];
#pragma unroll
    for (int kt = 0; kt < 4; ++kt) {
      const int c0 = kt * 32 + laneoff;
#pragma unroll
      for (int rr = 0; rr < 2; ++rr) {
        const int base = c0 + rr * 16;
        const float* s = (base < 64) ? (grow + base) : (crow + (base - 64));
#pragma unroll
        for (int e = 0; e < 8; ++e) a0[kt][rr * 8 + e] = (__bf16)s[e];
      }
    }
#pragma unroll
    for (int nt = 0; nt < 4; ++nt) {
      v8f acc = {};
#pragma unroll
      for (int kt = 0; kt < 4; ++kt) {
        v16bf bm = *(const v16bf*)&wfrag[((kt * 4 + nt) * 32 + lane) * 16];
        acc = __builtin_amdgcn_wmma_f32_16x16x32_bf16(
            false, a0[kt], false, bm, (short)0, acc, false, false);
      }
      const float bn = biasLds[0][nt * 16 + m];
#pragma unroll
      for (int i = 0; i < 8; ++i) {  // C layout: VGPR i -> row i + 8*halfw
        float v = acc[i] + bn; v = v > 0.f ? v : 0.f;
        h[(i + 8 * halfw) * 64 + nt * 16 + m] = (__bf16)v;
      }
    }

    // ---- Layers 1,2: [16x64] @ [64x64], activations via per-wave LDS h-buf.
#pragma unroll
    for (int layer = 1; layer <= 2; ++layer) {
      v16bf a[2];
#pragma unroll
      for (int kt = 0; kt < 2; ++kt) {
        const int c0 = kt * 32 + laneoff;
#pragma unroll
        for (int e = 0; e < 8; ++e) {
          a[kt][e]     = h[m * 64 + c0 + e];
          a[kt][8 + e] = h[m * 64 + c0 + 16 + e];
        }
      }
      const int fb = (layer == 1) ? 16 : 24;
#pragma unroll
      for (int nt = 0; nt < 4; ++nt) {
        v8f acc = {};
#pragma unroll
        for (int kt = 0; kt < 2; ++kt) {
          v16bf bm = *(const v16bf*)&wfrag[((fb + kt * 4 + nt) * 32 + lane) * 16];
          acc = __builtin_amdgcn_wmma_f32_16x16x32_bf16(
              false, a[kt], false, bm, (short)0, acc, false, false);
        }
        const float bn = biasLds[layer][nt * 16 + m];
#pragma unroll
        for (int i = 0; i < 8; ++i) {
          float v = acc[i] + bn; v = v > 0.f ? v : 0.f;
          h[(i + 8 * halfw) * 64 + nt * 16 + m] = (__bf16)v;
        }
      }
    }

    // ---- Final projection H->1 (VALU dot + cross-half shuffle reduce).
    float s = 0.f;
#pragma unroll
    for (int c = 0; c < 32; ++c)
      s += (float)h[m * 64 + halfw * 32 + c] * wfLds[halfw * 32 + c];
    s += __shfl_xor(s, 16);
    if (halfw == 0) logits[r0 + m] = s + bfLds;
  }
}

// ---------------------------------------------------------------------------
// Segment softmax + Gumbel argmax helpers
// ---------------------------------------------------------------------------
__device__ __forceinline__ unsigned ord_f32(float f) {
  unsigned u = __float_as_uint(f);
  return (u & 0x80000000u) ? ~u : (u | 0x80000000u);
}
__device__ __forceinline__ float unord_f32(unsigned u) {
  u = (u & 0x80000000u) ? (u & 0x7FFFFFFFu) : ~u;
  return __uint_as_float(u);
}
__device__ __forceinline__ unsigned hash_u32(unsigned x) {
  x ^= x >> 16; x *= 0x7FEB352Du;
  x ^= x >> 15; x *= 0x846CA68Bu;
  x ^= x >> 16;
  return x;
}

__global__ void seg_init_k(unsigned* segmax, float* segsum, int* counts,
                           unsigned long long* best, int B) {
  int b = blockIdx.x * blockDim.x + threadIdx.x;
  if (b < B) { segmax[b] = 0u; segsum[b] = 0.f; counts[b] = 0; best[b] = 0ull; }
}

__global__ void seg_max_k(const float* __restrict__ logits,
                          const int* __restrict__ bidx,
                          unsigned* __restrict__ segmax, int N) {
  int i = blockIdx.x * blockDim.x + threadIdx.x;
  if (i < N) atomicMax(&segmax[bidx[i]], ord_f32(logits[i]));
}

__global__ void seg_expsum_k(const float* __restrict__ logits,
                             const int* __restrict__ bidx,
                             const unsigned* __restrict__ segmax,
                             float* __restrict__ ews,
                             float* __restrict__ segsum,
                             int* __restrict__ counts, int N) {
  int i = blockIdx.x * blockDim.x + threadIdx.x;
  if (i < N) {
    int b = bidx[i];
    float e = __expf(logits[i] - unord_f32(segmax[b]));
    ews[i] = e;
    atomicAdd(&segsum[b], e);
    atomicAdd(&counts[b], 1);
  }
}

__global__ void scan_k(const int* __restrict__ counts, int* __restrict__ shift,
                       int B) {
  __shared__ int s[1024];
  int t = threadIdx.x;
  int c = (t < B) ? counts[t] : 0;
  s[t] = c;
  __syncthreads();
  for (int off = 1; off < 1024; off <<= 1) {
    int v = (t >= off) ? s[t - off] : 0;
    __syncthreads();
    s[t] += v;
    __syncthreads();
  }
  if (t < B) shift[t] = s[t] - c;  // exclusive prefix
}

__global__ void probs_pert_k(const float* __restrict__ ews,
                             const int* __restrict__ bidx,
                             const float* __restrict__ segsum,
                             float* __restrict__ out_probs,
                             unsigned long long* __restrict__ best, int N) {
  int i = blockIdx.x * blockDim.x + threadIdx.x;
  if (i < N) {
    int b = bidx[i];
    float p = ews[i] / (segsum[b] + EPS_F);
    out_probs[i] = p;
    if (p > EPS_F) {
      float lp = __logf(p);
      float u  = (float)hash_u32((unsigned)i) * 2.3283064e-10f + 1e-10f;
      float g  = -__logf(-__logf(u) + 1e-20f);  // Gumbel(0,1) via hash
      unsigned long long key =
          ((unsigned long long)ord_f32(lp + g) << 32) |
          (unsigned long long)(0xFFFFFFFFu - (unsigned)i);  // min-idx tiebreak
      atomicMax(&best[b], key);
    }
  }
}

__global__ void finalize_k(const unsigned long long* __restrict__ best,
                           const int* __restrict__ shift,
                           const float* __restrict__ probs,
                           float* __restrict__ out_alogp,
                           float* __restrict__ out_act,
                           float* __restrict__ out_shifted, int B) {
  int b = blockIdx.x * blockDim.x + threadIdx.x;
  if (b < B) {
    unsigned long long key = best[b];
    unsigned shifted = key ? (0xFFFFFFFFu - (unsigned)(key & 0xFFFFFFFFull))
                           : (unsigned)shift[b];
    float p = probs[shifted];
    out_alogp[b]   = __logf(p > 1e-30f ? p : 1e-30f);
    out_act[b]     = (float)((int)shifted - shift[b]);
    out_shifted[b] = (float)shifted;
  }
}

// ---------------------------------------------------------------------------
extern "C" void kernel_launch(void* const* d_in, const int* in_sizes, int n_in,
                              void* d_out, int out_size, void* d_ws,
                              size_t ws_size, hipStream_t stream) {
  (void)n_in; (void)out_size; (void)ws_size;
  const float* gemb = (const float*)d_in[0];
  const float* cand = (const float*)d_in[1];
  const int*   bidx = (const int*)d_in[2];
  const float* W0 = (const float*)d_in[3];  const float* b0 = (const float*)d_in[4];
  const float* W1 = (const float*)d_in[5];  const float* b1 = (const float*)d_in[6];
  const float* W2 = (const float*)d_in[7];  const float* b2 = (const float*)d_in[8];
  const float* Wf = (const float*)d_in[9];  const float* bfp = (const float*)d_in[10];

  const int B = in_sizes[0] / 64;
  const int N = in_sizes[2];

  // Output layout: g_emb[B,64] | X_states[N,128] | probs[N] | alogp[B] | act[B] | shifted[B]
  float* out        = (float*)d_out;
  float* out_gemb   = out;
  float* outX       = out_gemb + (size_t)B * 64;
  float* out_probs  = outX + (size_t)N * 128;
  float* out_alogp  = out_probs + (size_t)N;
  float* out_act    = out_alogp + (size_t)B;
  float* out_shifted= out_act + (size_t)B;

  // Workspace
  char* ws = (char*)d_ws;
  float* logits            = (float*)ws;              ws += (size_t)N * 4;
  float* ews               = (float*)ws;              ws += (size_t)N * 4;
  unsigned long long* best = (unsigned long long*)ws; ws += (size_t)B * 8;
  unsigned* segmax         = (unsigned*)ws;           ws += (size_t)B * 4;
  float* segsum            = (float*)ws;              ws += (size_t)B * 4;
  int* counts              = (int*)ws;                ws += (size_t)B * 4;
  int* shift               = (int*)ws;

  hipMemcpyAsync(out_gemb, gemb, (size_t)B * 64 * sizeof(float),
                 hipMemcpyDeviceToDevice, stream);

  const long nTiles = (long)N / 16;
  int blocks = (int)((nTiles + WAVES - 1) / WAVES);
  if (blocks > 8192) blocks = 8192;
  mlp_wmma_kernel<<<blocks, 256, 0, stream>>>(gemb, cand, bidx, W0, b0, W1, b1,
                                              W2, b2, Wf, bfp, outX, logits, N);

  seg_init_k<<<(B + 255) / 256, 256, 0, stream>>>(segmax, segsum, counts, best, B);
  seg_max_k<<<(N + 255) / 256, 256, 0, stream>>>(logits, bidx, segmax, N);
  seg_expsum_k<<<(N + 255) / 256, 256, 0, stream>>>(logits, bidx, segmax, ews,
                                                    segsum, counts, N);
  scan_k<<<1, 1024, 0, stream>>>(counts, shift, B);
  probs_pert_k<<<(N + 255) / 256, 256, 0, stream>>>(ews, bidx, segsum,
                                                    out_probs, best, N);
  finalize_k<<<(B + 255) / 256, 256, 0, stream>>>(best, shift, out_probs,
                                                  out_alogp, out_act,
                                                  out_shifted, B);
}
